// GaussianKernelLayer_46033459478723
// MI455X (gfx1250) — matrix operand order repaired and checked by reading, not compiled, not run on hardware
//
#include <hip/hip_runtime.h>
#include <stdint.h>

// CDNA5 / gfx1250: wave32, WMMA 16x16x32 bf16 -> f32.
typedef __attribute__((ext_vector_type(16))) __bf16 v16bf;
typedef __attribute__((ext_vector_type(8)))  float  v8f;

#define TPB    40                 // LDS t-stride in bf16 (80B: 16B-aligned, bank-spreading)
#define PL     (64 * TPB)         // one plane: 64 rows x TPB bf16 = 2560 bf16 (5120 B)
#define NPL    8                  // planes per buffer: 4 f x {hi, lo}
#define BUF16  (NPL * PL)         // 20480 bf16 = 40960 B per stage buffer
#define GP     68                 // Gram row stride (floats) for the epilogue scratch
#define NCH    32                 // 32 K-chunks of 32 cover T=1000 (zero padded to 1024)

union Frag {                       // v16bf assembled from two ds_load_b128
  v16bf v;
  uint4 q[2];
};
union Pack2 {                      // two bf16 packed into one dword store
  __bf16 h[2];
  uint32_t u;
};

__global__ __launch_bounds__(512)
void gaussian_gram_kernel(const float* __restrict__ x, float* __restrict__ out) {
  // Dynamic LDS: 2 stage buffers of bf16 hi/lo planes (81920 B),
  // reused as float G[4][64][GP] (69632 B) in the epilogue.
  extern __shared__ char smem[];
  __bf16* lds16 = reinterpret_cast<__bf16*>(smem);
  float*  ldsf  = reinterpret_cast<float*>(smem);

  const int tid  = threadIdx.x;
  const int lane = tid & 31;
  const int wave = tid >> 5;         // 16 waves
  const int n    = blockIdx.y;       // batch
  const int f0   = blockIdx.x * 4;   // feature group of 4

  const int f_l  = wave & 3;         // this wave's f within the group
  const int ti   = wave >> 2;        // this wave's 16-row tile strip (0..3)
  const int m    = lane & 15;
  const int half = lane >> 4;        // lane half per WMMA VGPR layout

  const float* xn = x + (size_t)n * 64 * 1000 * 16 + f0;  // + c*16000 + t*16

  // ---- cooperative stage: split f32 -> bf16 hi/lo ONCE per element, store packed.
  // Each thread owns 2 (c, t-pair) jobs per chunk: 64c x 16 pairs = 1024 jobs.
  auto load_chunk = [&](int buf, int chunk) {
    __bf16* base = lds16 + buf * BUF16;
#pragma unroll
    for (int it = 0; it < 2; ++it) {
      const int job = tid + it * 512;   // 0..1023
      const int c   = job >> 4;         // 0..63
      const int tp  = job & 15;         // t-pair index within chunk
      const int t   = chunk * 32 + tp * 2;   // t is even; t+1 <= 999 whenever t < 1000
      float4 v0 = make_float4(0.f, 0.f, 0.f, 0.f);
      float4 v1 = v0;
      if (t < 1000) {
        const float* g = xn + ((size_t)c * 1000 + t) * 16;
        v0 = *reinterpret_cast<const float4*>(g);
        v1 = *reinterpret_cast<const float4*>(g + 16);
        if (t + 64 < 1000)
          __builtin_prefetch(g + 64 * 16, 0, 1);   // global_prefetch_b8, 2 chunks ahead
      }
      const float a0[4] = {v0.x, v0.y, v0.z, v0.w};
      const float a1[4] = {v1.x, v1.y, v1.z, v1.w};
#pragma unroll
      for (int f = 0; f < 4; ++f) {
        __bf16 h0 = (__bf16)a0[f];             // pairs -> v_cvt_pk_bf16_f32
        __bf16 h1 = (__bf16)a1[f];
        float  r0 = a0[f] - (float)h0;         // exact residual
        float  r1 = a1[f] - (float)h1;
        Pack2 ph, pl2;
        ph.h[0]  = h0;        ph.h[1]  = h1;
        pl2.h[0] = (__bf16)r0; pl2.h[1] = (__bf16)r1;
        uint32_t* rowh = reinterpret_cast<uint32_t*>(base + (f * 2 + 0) * PL + c * TPB);
        uint32_t* rowl = reinterpret_cast<uint32_t*>(base + (f * 2 + 1) * PL + c * TPB);
        rowh[tp] = ph.u;                       // single ds_store_b32 per plane
        rowl[tp] = pl2.u;
      }
    }
  };

  v8f acc[4] = {};   // four 16x16 f32 tiles: (ti, j=0..3) of G for f = f_l

  load_chunk(0, 0);
  __syncthreads();

  int buf = 0;
  for (int ch = 0; ch < NCH; ++ch) {
    if (ch + 1 < NCH) load_chunk(buf ^ 1, ch + 1);   // double-buffered prefetch

    const __bf16* bh = lds16 + buf * BUF16 + (f_l * 2 + 0) * PL;   // hi plane of f
    const __bf16* bl = bh + PL;                                    // lo plane of f

    // ---- A fragments (16x32 bf16): lane half 0 holds K=0..7,16..23; half 1 K=8..15,24..31
    Frag Ah, Al;
    {
      const int ra = (16 * ti + m) * TPB + half * 8;
      Ah.q[0] = *reinterpret_cast<const uint4*>(bh + ra);
      Ah.q[1] = *reinterpret_cast<const uint4*>(bh + ra + 16);
      Al.q[0] = *reinterpret_cast<const uint4*>(bl + ra);
      Al.q[1] = *reinterpret_cast<const uint4*>(bl + ra + 16);
    }

#pragma unroll
    for (int j = 0; j < 4; ++j) {
      // ---- B fragments (32x16 bf16): lane (= column N) half 0 holds K=0..7,16..23; half 1 K=8..15,24..31
      const int rb = (16 * j + m) * TPB + half * 8;
      Frag Bh, Bl;
      Bh.q[0] = *reinterpret_cast<const uint4*>(bh + rb);
      Bh.q[1] = *reinterpret_cast<const uint4*>(bh + rb + 16);
      Bl.q[0] = *reinterpret_cast<const uint4*>(bl + rb);
      Bl.q[1] = *reinterpret_cast<const uint4*>(bl + rb + 16);

      // 3x bf16 split-product: Ah*Bh + Ah*Bl + Al*Bh  (~fp32 accuracy, 8x K of f32-WMMA)
      acc[j] = __builtin_amdgcn_wmma_f32_16x16x32_bf16(false, Ah.v, false, Bh.v,
                                                       (short)0, acc[j], false, false);
      acc[j] = __builtin_amdgcn_wmma_f32_16x16x32_bf16(false, Ah.v, false, Bl.v,
                                                       (short)0, acc[j], false, false);
      acc[j] = __builtin_amdgcn_wmma_f32_16x16x32_bf16(false, Al.v, false, Bh.v,
                                                       (short)0, acc[j], false, false);
    }

    __syncthreads();   // next-buffer writes + this-buffer reads complete
    buf ^= 1;
  }

  // ---- Spill Gram tiles to LDS (reuse stage memory): G[f][c][d], stride GP.
  // C/D layout: VGPR r, lanes 0-15 -> M=r, N=lane; lanes 16-31 -> M=r+8, N=lane-16.
#pragma unroll
  for (int j = 0; j < 4; ++j) {
#pragma unroll
    for (int r = 0; r < 8; ++r) {
      const int c = 16 * ti + r + half * 8;
      const int d = 16 * j + m;
      ldsf[(f_l * 64 + c) * GP + d] = acc[j][r];
    }
  }
  __syncthreads();

  // ---- Epilogue: d2 = G[c,c] + G[d,d] - 2 G[c,d]  (diagonal exactly 0 -> out 1.0)
#pragma unroll 4
  for (int rep = 0; rep < 32; ++rep) {
    const int e  = rep * 512 + tid;     // 0..16383 = 4f x 64c x 64d
    const int fl = e & 3;
    const int d  = (e >> 2) & 63;
    const int c  = e >> 8;
    const float* Gf = ldsf + fl * 64 * GP;
    const float gcc = Gf[c * GP + c];
    const float gdd = Gf[d * GP + d];
    const float gcd = Gf[c * GP + d];
    const float d2  = fmaxf(gcc + gdd - 2.0f * gcd, 0.0f);
    out[(((size_t)n * 64 + c) * 64 + d) * 16 + f0 + fl] = __expf(-0.5f * d2);
  }
}

extern "C" void kernel_launch(void* const* d_in, const int* in_sizes, int n_in,
                              void* d_out, int out_size, void* d_ws, size_t ws_size,
                              hipStream_t stream) {
  (void)in_sizes; (void)n_in; (void)d_ws; (void)ws_size; (void)out_size;
  const float* x = (const float*)d_in[0];
  float* out     = (float*)d_out;

  // Grid: 4 f-groups x 32 batches = 128 workgroups, 512 threads (16 wave32) each.
  dim3 grid(4, 32);
  dim3 block(512);
  const size_t shmem = 2 * BUF16 * sizeof(__bf16);   // 81920 B of the 320 KB WGP LDS
  hipLaunchKernelGGL(gaussian_gram_kernel, grid, block, shmem, stream, x, out);
}
